// EnBaseLayer_55216099557801
// MI455X (gfx1250) — matrix operand order repaired and checked by reading, not compiled, not run on hardware
//
#include <hip/hip_runtime.h>
#include <hip/hip_bf16.h>

#define HID 128
#define EF  16

typedef __attribute__((ext_vector_type(16))) __bf16 v16bf;
typedef __attribute__((ext_vector_type(8)))  float  v8f;

union BF16x16 { v16bf v; unsigned int u[8]; uint4 q[2]; };

__device__ __forceinline__ unsigned short f2bf(float f) {
  unsigned int x = __float_as_uint(f);
  x += 0x7FFFu + ((x >> 16) & 1u);          // round-to-nearest-even
  return (unsigned short)(x >> 16);
}

__device__ __forceinline__ float silu_f(float v) {
  return v / (1.0f + __expf(-v));
}

__device__ __forceinline__ v8f wmma_bf16(const BF16x16& a, const BF16x16& b, v8f c) {
  return __builtin_amdgcn_wmma_f32_16x16x32_bf16(false, a.v, false, b.v,
                                                 (short)0, c, false, false);
}

// ---------------------------------------------------------------------------
// Repack a row-major f32 weight W[kdim x 128] into per-wave bf16 B-operand
// layout: dst[((kc*8 + ntile)*32 + lane)*8 + v] packs K = kc*32 + (v&4?16:0)
// + (lane&16?8:0) + 2(v&3), +1 for column n = ntile*16 + (lane&15). K zero-padded past kdim.
// ---------------------------------------------------------------------------
__global__ void pack_w(const float* __restrict__ w, unsigned int* __restrict__ dst,
                       int kdim, int nkc) {
  int idx = blockIdx.x * blockDim.x + threadIdx.x;
  if (idx >= nkc * 2048) return;
  int v    = idx & 7;
  int lane = (idx >> 3) & 31;
  int nt   = (idx >> 8) & 7;
  int kc   = idx >> 11;
  int n  = nt * 16 + (lane & 15);
  int k0 = kc * 32 + ((v & 4) ? 16 : 0) + ((lane & 16) ? 8 : 0) + 2 * (v & 3);
  float w0 = (k0     < kdim) ? w[(size_t)k0       * HID + n] : 0.0f;
  float w1 = (k0 + 1 < kdim) ? w[(size_t)(k0 + 1) * HID + n] : 0.0f;
  dst[idx] = (unsigned int)f2bf(w0) | ((unsigned int)f2bf(w1) << 16);
}

// Zero mi accumulator and seed x_out = x.
__global__ void init_buffers(float* __restrict__ mi_acc, float* __restrict__ x_out,
                             const float* __restrict__ x, int n_nodes) {
  int i = blockIdx.x * 256 + threadIdx.x;
  int nmi = n_nodes * HID;
  if (i < nmi)                       mi_acc[i] = 0.0f;
  else if (i < nmi + n_nodes * 3)    x_out[i - nmi] = x[i - nmi];
}

// ---------------------------------------------------------------------------
// Edge kernel: 16 edges per workgroup, 8 waves; wave w owns columns [16w,16w+16).
// ---------------------------------------------------------------------------
__global__ __launch_bounds__(256) void egnn_edge_kernel(
    const float* __restrict__ h, const float* __restrict__ x,
    const int* __restrict__ eidx, const float* __restrict__ eattr,
    const unsigned int* __restrict__ em_w1p, const float* __restrict__ em_b1,
    const unsigned int* __restrict__ em_w2p, const float* __restrict__ em_b2,
    const float* __restrict__ ei_w, const float* __restrict__ ei_b,
    const unsigned int* __restrict__ xm_w1p, const float* __restrict__ xm_b1,
    const float* __restrict__ xm_w2,
    float* __restrict__ mi_acc, float* __restrict__ x_out, int n_edges)
{
  __shared__ __attribute__((aligned(16))) unsigned short sA[16][288]; // edge_in bf16
  __shared__ __attribute__((aligned(16))) unsigned short sT[16][128]; // act bf16
  __shared__ __attribute__((aligned(16))) float sM[16][128];          // mij f32
  __shared__ float sRed[16];
  __shared__ float sGate[16];
  __shared__ int   sDst[16];
  __shared__ float sRel[16][3];
  __shared__ float sInv[16];

  const int tid  = threadIdx.x;
  const int wave = tid >> 5;
  const int lane = tid & 31;
  const int base = blockIdx.x * 16;

  // ---- Phase 0: gather edge inputs -> LDS (bf16) ----
  {
    int e   = tid >> 4;
    int c16 = tid & 15;
    int eg  = base + e; if (eg >= n_edges) eg = n_edges - 1;
    int si  = eidx[eg];               // src
    int di  = eidx[n_edges + eg];     // dst
    const float* hi = h + (size_t)di * HID;
    const float* hj = h + (size_t)si * HID;
    #pragma unroll
    for (int c = c16; c < HID; c += 16) {
      sA[e][c]       = f2bf(hi[c]);
      sA[e][HID + c] = f2bf(hj[c]);
    }
    sA[e][2 * HID + 1 + c16] = f2bf(eattr[(size_t)eg * EF + c16]);
    if (c16 < 15) sA[e][273 + c16] = 0;           // pad K 273..287
    if (c16 == 0) {
      float rx = x[(size_t)di * 3 + 0] - x[(size_t)si * 3 + 0];
      float ry = x[(size_t)di * 3 + 1] - x[(size_t)si * 3 + 1];
      float rz = x[(size_t)di * 3 + 2] - x[(size_t)si * 3 + 2];
      float dsq = rx * rx + ry * ry + rz * rz;
      sA[e][2 * HID] = f2bf(dsq);
      sDst[e] = di;
      sRel[e][0] = rx; sRel[e][1] = ry; sRel[e][2] = rz;
      sInv[e] = 1.0f / (sqrtf(dsq + 1e-8f) + 1.0f);
    }
  }
  __syncthreads();

  const int colq  = lane & 15;
  const int col   = wave * 16 + colq;
  const int khalf = (lane & 16) ? 8 : 0;   // A-layout K half offset (ushorts)
  const int mhalf = (lane & 16) ? 8 : 0;   // C-layout row half

  // ---- Phase 1: layer 1 (K=288 -> 9 chunks) ----
  v8f acc = {};
  #pragma unroll
  for (int kc = 0; kc < 9; ++kc) {
    BF16x16 a, b;
    a.q[0] = *(const uint4*)(&sA[colq][kc * 32 + khalf]);
    a.q[1] = *(const uint4*)(&sA[colq][kc * 32 + 16 + khalf]);
    const uint4* bp = (const uint4*)(em_w1p + ((size_t)(kc * 8 + wave) * 32 + lane) * 8);
    b.q[0] = bp[0]; b.q[1] = bp[1];
    acc = wmma_bf16(a, b, acc);
  }
  {
    float bias = em_b1[col];
    #pragma unroll
    for (int r = 0; r < 8; ++r)
      sT[r + mhalf][col] = f2bf(silu_f(acc[r] + bias));
  }
  __syncthreads();

  // ---- Phase 2: layer 2 (K=128 -> 4 chunks) -> mij ----
  v8f acc2 = {};
  #pragma unroll
  for (int kc = 0; kc < 4; ++kc) {
    BF16x16 a, b;
    a.q[0] = *(const uint4*)(&sT[colq][kc * 32 + khalf]);
    a.q[1] = *(const uint4*)(&sT[colq][kc * 32 + 16 + khalf]);
    const uint4* bp = (const uint4*)(em_w2p + ((size_t)(kc * 8 + wave) * 32 + lane) * 8);
    b.q[0] = bp[0]; b.q[1] = bp[1];
    acc2 = wmma_bf16(a, b, acc2);
  }
  __syncthreads();                       // done reading sT before overwrite
  {
    float bias = em_b2[col];
    #pragma unroll
    for (int r = 0; r < 8; ++r) {
      float m = acc2[r] + bias;
      sM[r + mhalf][col] = m;            // f32 copy for reductions/scatter
      sT[r + mhalf][col] = f2bf(m);      // bf16 copy as next A operand
    }
  }
  if (tid < 16) { sRed[tid] = 0.0f; sGate[tid] = 0.0f; }
  __syncthreads();

  // ---- Phase 3: eij partial reduction + x-gate MLP ----
  {
    int e = tid >> 4, c0 = tid & 15;
    float p = 0.0f;
    #pragma unroll
    for (int c = c0; c < HID; c += 16) p += sM[e][c] * ei_w[c];
    atomicAdd(&sRed[e], p);
  }
  v8f acc3 = {};
  #pragma unroll
  for (int kc = 0; kc < 4; ++kc) {
    BF16x16 a, b;
    a.q[0] = *(const uint4*)(&sT[colq][kc * 32 + khalf]);
    a.q[1] = *(const uint4*)(&sT[colq][kc * 32 + 16 + khalf]);
    const uint4* bp = (const uint4*)(xm_w1p + ((size_t)(kc * 8 + wave) * 32 + lane) * 8);
    b.q[0] = bp[0]; b.q[1] = bp[1];
    acc3 = wmma_bf16(a, b, acc3);
  }
  {
    float bias = xm_b1[col];
    float wcol = xm_w2[col];
    #pragma unroll
    for (int r = 0; r < 8; ++r)
      atomicAdd(&sGate[r + mhalf], silu_f(acc3[r] + bias) * wcol);
  }
  __syncthreads();

  // ---- Phase 4: scatter outputs ----
  float eib = ei_b[0];
  if (tid < 16) {
    int e = tid;
    if (base + e < n_edges) {
      float s = sInv[e] * tanhf(sGate[e]);
      atomicAdd(&x_out[(size_t)sDst[e] * 3 + 0], sRel[e][0] * s);
      atomicAdd(&x_out[(size_t)sDst[e] * 3 + 1], sRel[e][1] * s);
      atomicAdd(&x_out[(size_t)sDst[e] * 3 + 2], sRel[e][2] * s);
    }
  }
  {
    int e = tid >> 4, c0 = tid & 15;
    if (base + e < n_edges) {
      float eij = 1.0f / (1.0f + __expf(-(sRed[e] + eib)));
      float* dp = mi_acc + (size_t)sDst[e] * HID;
      #pragma unroll
      for (int c = c0; c < HID; c += 16)
        atomicAdd(&dp[c], sM[e][c] * eij);
    }
  }
}

// ---------------------------------------------------------------------------
// Node kernel: h_out = h + MLP(concat(mi, h))
// ---------------------------------------------------------------------------
__global__ __launch_bounds__(256) void egnn_node_kernel(
    const float* __restrict__ h, const float* __restrict__ mi_acc,
    const unsigned int* __restrict__ nm_w1p, const float* __restrict__ nm_b1,
    const unsigned int* __restrict__ nm_w2p, const float* __restrict__ nm_b2,
    float* __restrict__ h_out, int n_nodes)
{
  __shared__ __attribute__((aligned(16))) unsigned short sA[16][256];
  __shared__ __attribute__((aligned(16))) unsigned short sT[16][128];

  const int tid  = threadIdx.x;
  const int wave = tid >> 5;
  const int lane = tid & 31;
  const int base = blockIdx.x * 16;

  {
    int r = tid >> 4, c16 = tid & 15;
    int n = base + r; if (n >= n_nodes) n = n_nodes - 1;
    const float* mp = mi_acc + (size_t)n * HID;
    const float* hp = h + (size_t)n * HID;
    #pragma unroll
    for (int c = c16; c < HID; c += 16) {
      sA[r][c]       = f2bf(mp[c]);
      sA[r][HID + c] = f2bf(hp[c]);
    }
  }
  __syncthreads();

  const int colq  = lane & 15;
  const int col   = wave * 16 + colq;
  const int khalf = (lane & 16) ? 8 : 0;
  const int mhalf = (lane & 16) ? 8 : 0;

  v8f acc = {};
  #pragma unroll
  for (int kc = 0; kc < 8; ++kc) {
    BF16x16 a, b;
    a.q[0] = *(const uint4*)(&sA[colq][kc * 32 + khalf]);
    a.q[1] = *(const uint4*)(&sA[colq][kc * 32 + 16 + khalf]);
    const uint4* bp = (const uint4*)(nm_w1p + ((size_t)(kc * 8 + wave) * 32 + lane) * 8);
    b.q[0] = bp[0]; b.q[1] = bp[1];
    acc = wmma_bf16(a, b, acc);
  }
  {
    float bias = nm_b1[col];
    #pragma unroll
    for (int r = 0; r < 8; ++r)
      sT[r + mhalf][col] = f2bf(silu_f(acc[r] + bias));
  }
  __syncthreads();

  v8f acc2 = {};
  #pragma unroll
  for (int kc = 0; kc < 4; ++kc) {
    BF16x16 a, b;
    a.q[0] = *(const uint4*)(&sT[colq][kc * 32 + khalf]);
    a.q[1] = *(const uint4*)(&sT[colq][kc * 32 + 16 + khalf]);
    const uint4* bp = (const uint4*)(nm_w2p + ((size_t)(kc * 8 + wave) * 32 + lane) * 8);
    b.q[0] = bp[0]; b.q[1] = bp[1];
    acc2 = wmma_bf16(a, b, acc2);
  }
  {
    float bias = nm_b2[col];
    #pragma unroll
    for (int r = 0; r < 8; ++r) {
      int n = base + r + mhalf;
      if (n < n_nodes)
        h_out[(size_t)n * HID + col] = h[(size_t)n * HID + col] + acc2[r] + bias;
    }
  }
}

// ---------------------------------------------------------------------------
extern "C" void kernel_launch(void* const* d_in, const int* in_sizes, int n_in,
                              void* d_out, int out_size, void* d_ws, size_t ws_size,
                              hipStream_t stream) {
  const float* h         = (const float*)d_in[0];
  const float* x         = (const float*)d_in[1];
  const int*   edge_index= (const int*)  d_in[2];
  const float* edge_attr = (const float*)d_in[3];
  const float* em_w1     = (const float*)d_in[4];
  const float* em_b1     = (const float*)d_in[5];
  const float* em_w2     = (const float*)d_in[6];
  const float* em_b2     = (const float*)d_in[7];
  const float* ei_w      = (const float*)d_in[8];
  const float* ei_b      = (const float*)d_in[9];
  const float* xm_w1     = (const float*)d_in[10];
  const float* xm_b1     = (const float*)d_in[11];
  const float* xm_w2     = (const float*)d_in[12];
  const float* nm_w1     = (const float*)d_in[13];
  const float* nm_b1     = (const float*)d_in[14];
  const float* nm_w2     = (const float*)d_in[15];
  const float* nm_b2     = (const float*)d_in[16];

  const int n_nodes = in_sizes[0] / HID;   // 50000
  const int n_edges = in_sizes[2] / 2;     // 640000

  float* h_out = (float*)d_out;
  float* x_out = (float*)d_out + (size_t)n_nodes * HID;

  // Workspace layout
  char* ws = (char*)d_ws;
  float* mi_acc = (float*)ws;
  size_t off = (size_t)n_nodes * HID * sizeof(float);
  unsigned int* em_w1p = (unsigned int*)(ws + off); off += (size_t)9 * 2048 * 4;
  unsigned int* em_w2p = (unsigned int*)(ws + off); off += (size_t)4 * 2048 * 4;
  unsigned int* xm_w1p = (unsigned int*)(ws + off); off += (size_t)4 * 2048 * 4;
  unsigned int* nm_w1p = (unsigned int*)(ws + off); off += (size_t)8 * 2048 * 4;
  unsigned int* nm_w2p = (unsigned int*)(ws + off); off += (size_t)4 * 2048 * 4;

  // Init: zero mi_acc, seed x_out = x
  {
    int n = n_nodes * HID + n_nodes * 3;
    init_buffers<<<(n + 255) / 256, 256, 0, stream>>>(mi_acc, x_out, x, n_nodes);
  }
  // Pack weights to bf16 WMMA B-operand layout
  pack_w<<<(9 * 2048 + 255) / 256, 256, 0, stream>>>(em_w1, em_w1p, 273, 9);
  pack_w<<<(4 * 2048 + 255) / 256, 256, 0, stream>>>(em_w2, em_w2p, HID, 4);
  pack_w<<<(4 * 2048 + 255) / 256, 256, 0, stream>>>(xm_w1, xm_w1p, HID, 4);
  pack_w<<<(8 * 2048 + 255) / 256, 256, 0, stream>>>(nm_w1, nm_w1p, 2 * HID, 8);
  pack_w<<<(4 * 2048 + 255) / 256, 256, 0, stream>>>(nm_w2, nm_w2p, HID, 4);

  // Fused edge pipeline (WMMA) + scatter
  egnn_edge_kernel<<<(n_edges + 15) / 16, 256, 0, stream>>>(
      h, x, edge_index, edge_attr,
      em_w1p, em_b1, em_w2p, em_b2,
      ei_w, ei_b, xm_w1p, xm_b1, xm_w2,
      mi_acc, x_out, n_edges);

  // Node MLP (WMMA)
  egnn_node_kernel<<<(n_nodes + 15) / 16, 256, 0, stream>>>(
      h, mi_acc, nm_w1p, nm_b1, nm_w2p, nm_b2, h_out, n_nodes);
}
